// MultimodalBaselineMoELayer_68023692034429
// MI455X (gfx1250) — compile-verified, matrix operand order checked by reading, not run on hardware
//
#include <hip/hip_runtime.h>
#include <hip/hip_bf16.h>

// ---------------------------------------------------------------------------
// MoE layer for MI455X (gfx1250): bf16 WMMA fused expert FFN with top-2 routing
// Weights pre-packed into CDNA5 B-fragment layout (bf16) once per launch.
// 32-token tiles: every B fragment feeds two WMMA (2 M-tiles) -> half the
// weight traffic per FLOP vs a 16-row tile.
// ---------------------------------------------------------------------------
#define B_  4
#define M_  3
#define T_  2048
#define D_  1024
#define E_  16
#define H_  2048
#define N_  (B_ * M_ * T_)     // 24576 tokens

#define MT_   32               // tokens per block (2 WMMA M-tiles)
#define HC_   256              // H chunk held in LDS per pass
#define XPAD  8
#define HPAD  8

typedef __attribute__((ext_vector_type(16))) __bf16 v16bf;
typedef __attribute__((ext_vector_type(8)))  __bf16 v8bf;
typedef __attribute__((ext_vector_type(8)))  float  v8f;

union BF16x16 { v16bf v; v8bf h[2]; __bf16 e[16]; };

// ---------------------------------------------------------------------------
// Kernel 0: pack a row-major K x Ncols fp32 matrix (per expert) into bf16
// B-fragments.  Fragment block (e, nt, kb) = 16x32 tile of columns
// [nt*16, nt*16+16) and rows [kb*32, kb*32+32), stored as 32 lanes x 16 bf16
// in the exact V_WMMA_16X16X32_BF16 B operand lane layout:
//   lane<16 : col = nt*16+lane,    K = kb*32 + {0..7, 16..23}
//   lane>=16: col = nt*16+lane-16, K = kb*32 + {8..15, 24..31}
// ---------------------------------------------------------------------------
__global__ void __launch_bounds__(256)
moe_pack(const float* __restrict__ W, __bf16* __restrict__ P,
         int Ncols, int KB, int NT) {
    const int idx  = blockIdx.x * 256 + threadIdx.x;
    const int lane = idx & 31;
    const int fb   = idx >> 5;             // (e*NT + nt)*KB + kb
    const int kb   = fb % KB;
    const int nt   = (fb / KB) % NT;
    const int e    = fb / (KB * NT);
    const int koff = (lane >> 4) * 8;
    const int col  = nt * 16 + (lane & 15);

    const float* src = W + (size_t)e * (size_t)(KB * 32) * Ncols
                         + (size_t)(kb * 32 + koff) * Ncols + col;
    BF16x16 u;
    #pragma unroll
    for (int j = 0; j < 8; ++j) {
        u.e[j]     = (__bf16)src[(size_t)j * Ncols];
        u.e[8 + j] = (__bf16)src[(size_t)(16 + j) * Ncols];
    }
    v8bf* dst = (v8bf*)(P + ((size_t)fb * 32 + lane) * 16);
    dst[0] = u.h[0];
    dst[1] = u.h[1];
}

// ---------------------------------------------------------------------------
// Kernel 1: router.  One wave32 per token; lanes 0..15 = experts, the two
// 16-lane halves split the D reduction.  Lane 0 picks top-2, softmaxes, and
// atomically appends (token, weight) to per-expert lists in workspace.
// ---------------------------------------------------------------------------
__global__ void __launch_bounds__(256)
moe_gate(const float* __restrict__ f0, const float* __restrict__ f1,
         const float* __restrict__ f2, const float* __restrict__ gw,
         int* __restrict__ cnt, int* __restrict__ tokidx,
         float* __restrict__ tokw) {
    const int lane = threadIdx.x & 31;
    const int wid  = threadIdx.x >> 5;
    const int n    = blockIdx.x * 8 + wid;
    if (n >= N_) return;

    const int t  = n % T_;
    const int mm = (n / T_) % M_;
    const int b  = n / (T_ * M_);
    const float* xrow = (mm == 0 ? f0 : (mm == 1 ? f1 : f2)) +
                        (size_t)(b * T_ + t) * D_;

    const int e    = lane & 15;
    const int half = lane >> 4;
    const int d0   = half * (D_ / 2);
    float acc = 0.f;
    #pragma unroll 4
    for (int d = 0; d < D_ / 2; ++d)
        acc = fmaf(xrow[d0 + d], gw[(size_t)(d0 + d) * E_ + e], acc);
    acc += __shfl_down(acc, 16, 32);     // lanes 0..15 now hold full logits

    float v1 = -1e30f, v2 = -1e30f;
    int   i1 = 0, i2 = 0;
    #pragma unroll
    for (int ee = 0; ee < E_; ++ee) {
        float v = __shfl(acc, ee, 32);
        if (lane == 0) {
            if (v > v1)      { v2 = v1; i2 = i1; v1 = v; i1 = ee; }
            else if (v > v2) { v2 = v;  i2 = ee; }
        }
    }
    if (lane == 0) {
        float e2 = __expf(v2 - v1);
        float s  = 1.f + e2;
        float p1 = 1.f / s, p2 = e2 / s;
        int s1 = atomicAdd(&cnt[i1], 1);
        tokidx[i1 * N_ + s1] = n;  tokw[i1 * N_ + s1] = p1;
        int s2 = atomicAdd(&cnt[i2], 1);
        tokidx[i2 * N_ + s2] = n;  tokw[i2 * N_ + s2] = p2;
    }
}

// ---------------------------------------------------------------------------
// Kernel 2: fused expert FFN.  One block = 32 routed tokens of one expert.
// X tile cached in LDS (bf16); H chunk computed in LDS; Y in registers.
// B operands come pre-swizzled from packed bf16 weights: 2x16B loads/fragment,
// each reused by two WMMA (two M-tiles).
// ---------------------------------------------------------------------------
__global__ void __launch_bounds__(256, 1)
moe_ffn(const float* __restrict__ f0, const float* __restrict__ f1,
        const float* __restrict__ f2,
        const __bf16* __restrict__ w1p, const float* __restrict__ b1,
        const __bf16* __restrict__ w2p, const float* __restrict__ b2,
        const int* __restrict__ cnt, const int* __restrict__ tokidx,
        const float* __restrict__ tokw, float* __restrict__ out) {
    __shared__ __attribute__((aligned(16))) __bf16 ldsX[MT_][D_ + XPAD];
    __shared__ __attribute__((aligned(16))) __bf16 ldsH[MT_][HC_ + HPAD];
    __shared__ int   sTok[MT_];
    __shared__ float sW[MT_];
    __shared__ int   sOutBase[MT_];

    const int e  = blockIdx.y;
    const int cn = cnt[e];
    const int i0 = blockIdx.x * MT_;
    if (i0 >= cn) return;

    const int tid  = threadIdx.x;
    const int lane = tid & 31;
    const int wv   = tid >> 5;                 // wave 0..7

    if (tid < MT_) {
        int i = i0 + tid;
        int g = 0; float wgt = 0.f;
        if (i < cn) { g = tokidx[e * N_ + i]; wgt = tokw[e * N_ + i]; }
        sTok[tid] = g;
        sW[tid]   = wgt * (1.f / 3.f);         // fold modality mean
        int t = g % T_;
        int b = g / (T_ * M_);
        sOutBase[tid] = (b * T_ + t) * D_;
    }
    __syncthreads();

    // Gather MT_ token rows (fp32 -> bf16) into LDS, fully coalesced.
    for (int j = tid; j < MT_ * D_; j += 256) {
        int r = j >> 10;
        int c = j & (D_ - 1);
        int g  = sTok[r];
        int t  = g % T_;
        int mm = (g / T_) % M_;
        int b  = g / (T_ * M_);
        const float* base = (mm == 0 ? f0 : (mm == 1 ? f1 : f2));
        ldsX[r][c] = (__bf16)base[(size_t)(b * T_ + t) * D_ + c];
    }
    __syncthreads();

    // packed weight bases:  w1p: NT1=H/16=128 tiles, KB1=D/32=32 blocks
    //                       w2p: NT2=D/16=64  tiles, KB2=H/32=64 blocks
    const __bf16* w1e = w1p + (size_t)e * ((size_t)(H_ / 16) * (D_ / 32) * 512);
    const __bf16* w2e = w2p + (size_t)e * ((size_t)(D_ / 16) * (H_ / 32) * 512);
    const float*  b1e = b1 + e * H_;
    const float*  b2e = b2 + e * D_;

    const int koff    = (lane >> 4) * 8;       // CDNA5 16-bit A K lane split
    const int col16   = lane & 15;
    const int rowbase = (lane >> 4) * 8;       // C/D: VGPR r -> row r (+8)

    // Y accumulators: wave owns D columns [wv*128, wv*128+128) for both
    // M-tiles (rows 0..15 -> y0, rows 16..31 -> y1), init with b2.
    v8f y0[8], y1[8];
    #pragma unroll
    for (int yt = 0; yt < 8; ++yt) {
        float bv = b2e[wv * 128 + yt * 16 + col16];
        #pragma unroll
        for (int r = 0; r < 8; ++r) { y0[yt][r] = bv; y1[yt][r] = bv; }
    }
    // per-wave B column bases for phase 2 (8 output tiles)
    const __bf16* bcol2[8];
    #pragma unroll
    for (int yt = 0; yt < 8; ++yt)
        bcol2[yt] = w2e + ((size_t)(wv * 8 + yt) * (H_ / 32) * 32 + lane) * 16;

    for (int ch = 0; ch < H_ / HC_; ++ch) {
        __syncthreads();   // everyone done reading ldsH of previous chunk

        // ---- phase 1: Hc = relu(X @ W1[:, chunk] + b1) --------------------
        #pragma unroll
        for (int nt = 0; nt < 2; ++nt) {
            const int ht   = ch * (HC_ / 16) + wv * 2 + nt;  // global h-tile
            const int hcol = ht * 16 + col16;
            const __bf16* bcol = w1e + ((size_t)ht * (D_ / 32) * 32 + lane) * 16;

            v8f c0, c1;
            float b1v = b1e[hcol];
            #pragma unroll
            for (int r = 0; r < 8; ++r) { c0[r] = b1v; c1[r] = b1v; }

            #pragma unroll 2
            for (int ks = 0; ks < D_ / 32; ++ks) {
                const int k0 = ks * 32;
                BF16x16 a0, a1;
                a0.h[0] = *(const v8bf*)&ldsX[col16][k0 + koff];
                a0.h[1] = *(const v8bf*)&ldsX[col16][k0 + 16 + koff];
                a1.h[0] = *(const v8bf*)&ldsX[16 + col16][k0 + koff];
                a1.h[1] = *(const v8bf*)&ldsX[16 + col16][k0 + 16 + koff];
                BF16x16 bb;
                const v8bf* bp = (const v8bf*)(bcol + (size_t)ks * 512);
                bb.h[0] = bp[0];
                bb.h[1] = bp[1];
                if (ks + 1 < D_ / 32)
                    __builtin_prefetch(bcol + (size_t)(ks + 1) * 512, 0, 1);
                c0 = __builtin_amdgcn_wmma_f32_16x16x32_bf16(
                         false, a0.v, false, bb.v, (short)0, c0, false, false);
                c1 = __builtin_amdgcn_wmma_f32_16x16x32_bf16(
                         false, a1.v, false, bb.v, (short)0, c1, false, false);
            }
            const int cc = (wv * 2 + nt) * 16 + col16;
            #pragma unroll
            for (int r = 0; r < 8; ++r) {
                ldsH[rowbase + r][cc]      = (__bf16)fmaxf(c0[r], 0.f);
                ldsH[16 + rowbase + r][cc] = (__bf16)fmaxf(c1[r], 0.f);
            }
        }
        __syncthreads();

        // ---- phase 2: Y += Hc @ W2[chunk rows, :] -------------------------
        #pragma unroll 1
        for (int kk = 0; kk < HC_; kk += 32) {
            BF16x16 a0, a1;
            a0.h[0] = *(const v8bf*)&ldsH[col16][kk + koff];
            a0.h[1] = *(const v8bf*)&ldsH[col16][kk + 16 + koff];
            a1.h[0] = *(const v8bf*)&ldsH[16 + col16][kk + koff];
            a1.h[1] = *(const v8bf*)&ldsH[16 + col16][kk + 16 + koff];
            const int ks2 = (ch * HC_ + kk) >> 5;    // global H k-block
            #pragma unroll
            for (int yt = 0; yt < 8; ++yt) {
                BF16x16 bb;
                const v8bf* bp = (const v8bf*)(bcol2[yt] + (size_t)ks2 * 512);
                bb.h[0] = bp[0];
                bb.h[1] = bp[1];
                y0[yt] = __builtin_amdgcn_wmma_f32_16x16x32_bf16(
                             false, a0.v, false, bb.v, (short)0, y0[yt], false, false);
                y1[yt] = __builtin_amdgcn_wmma_f32_16x16x32_bf16(
                             false, a1.v, false, bb.v, (short)0, y1[yt], false, false);
            }
        }
    }

    // ---- scatter: out[b,t,:] += (cw/3) * y --------------------------------
    #pragma unroll
    for (int yt = 0; yt < 8; ++yt) {
        const int d = wv * 128 + yt * 16 + col16;
        #pragma unroll
        for (int r = 0; r < 8; ++r) {
            const int m0 = rowbase + r;
            const int m1 = 16 + rowbase + r;
            if (i0 + m0 < cn)
                atomicAdd(out + sOutBase[m0] + d, y0[yt][r] * sW[m0]);
            if (i0 + m1 < cn)
                atomicAdd(out + sOutBase[m1] + d, y1[yt][r] * sW[m1]);
        }
    }
}

// ---------------------------------------------------------------------------
extern "C" void kernel_launch(void* const* d_in, const int* in_sizes, int n_in,
                              void* d_out, int out_size, void* d_ws, size_t ws_size,
                              hipStream_t stream) {
    const float* f0 = (const float*)d_in[0];
    const float* f1 = (const float*)d_in[1];
    const float* f2 = (const float*)d_in[2];
    const float* gw = (const float*)d_in[3];
    const float* w1 = (const float*)d_in[4];
    const float* b1 = (const float*)d_in[5];
    const float* w2 = (const float*)d_in[6];
    const float* b2 = (const float*)d_in[7];
    float* out = (float*)d_out;

    // workspace layout (bytes):
    //   [0,256)                    cnt (E ints, padded)
    //   [256, +E*N*4)              tokidx
    //   [.., +E*N*4)               tokw
    //   [.., +E*D*H*2)             w1 packed bf16 fragments (67 MB)
    //   [.., +E*H*D*2)             w2 packed bf16 fragments (67 MB)
    char* ws = (char*)d_ws;
    int*    cnt    = (int*)ws;
    int*    tokidx = (int*)(ws + 256);
    float*  tokw   = (float*)(ws + 256 + (size_t)E_ * N_ * 4);
    __bf16* w1pk   = (__bf16*)(ws + 256 + (size_t)E_ * N_ * 8);
    __bf16* w2pk   = w1pk + (size_t)E_ * D_ * H_;

    hipMemsetAsync(cnt, 0, 64 * sizeof(int), stream);
    hipMemsetAsync(out, 0, (size_t)out_size * sizeof(float), stream);

    // pack weights into WMMA B-fragment layout (bf16)
    {
        // w1: K=D rows, Ncols=H  -> KB=D/32, NT=H/16
        int KB = D_ / 32, NT = H_ / 16;
        int tasks = E_ * NT * KB * 32;
        moe_pack<<<tasks / 256, 256, 0, stream>>>(w1, w1pk, H_, KB, NT);
    }
    {
        // w2: K=H rows, Ncols=D  -> KB=H/32, NT=D/16
        int KB = H_ / 32, NT = D_ / 16;
        int tasks = E_ * NT * KB * 32;
        moe_pack<<<tasks / 256, 256, 0, stream>>>(w2, w2pk, D_, KB, NT);
    }

    moe_gate<<<N_ / 8, 256, 0, stream>>>(f0, f1, f2, gw, cnt, tokidx, tokw);

    dim3 grid((N_ + MT_ - 1) / MT_, E_);
    moe_ffn<<<grid, 256, 0, stream>>>(f0, f1, f2, w1pk, b1, w2pk, b2,
                                      cnt, tokidx, tokw, out);
}
